// TransformerBlock_10995116278103
// MI455X (gfx1250) — compile-verified, matrix operand order checked by reading, not run on hardware
//
#include <hip/hip_runtime.h>

// ---------------------------------------------------------------------------
// Types for CDNA5 WMMA (wave32): v_wmma_f32_16x16x32_bf16
// ---------------------------------------------------------------------------
typedef __bf16 bf16;
typedef __attribute__((ext_vector_type(16))) __bf16 v16bf;
typedef __attribute__((ext_vector_type(8)))  __bf16 v8bf;
typedef __attribute__((ext_vector_type(8)))  float  v8f;

// D = A(16x32) * B(32x16) + C, bf16 in / f32 accum
__device__ __forceinline__ v8f wmma_bf(v16bf a, v16bf b, v8f c) {
    return __builtin_amdgcn_wmma_f32_16x16x32_bf16(
        /*neg_a=*/false, a, /*neg_b=*/false, b,
        /*c_mod=*/(short)0, c, /*reuse_a=*/false, /*reuse_b=*/false);
}

// Load one 16-element bf16 fragment (A or B layout) for this lane.
// Per ISA: lane holds 16 bf16 at K = (e&7) + 8*(lane>>4) + 16*(e>>3),
// i.e. two contiguous 8-element (16B) chunks at p and p+16.
__device__ __forceinline__ v16bf load_frag(const bf16* p) {
    v8bf lo = *reinterpret_cast<const v8bf*>(p);
    v8bf hi = *reinterpret_cast<const v8bf*>(p + 16);
    v16bf f;
#pragma unroll
    for (int i = 0; i < 8; ++i) { f[i] = lo[i]; f[i + 8] = hi[i]; }
    return f;
}

// ---------------------------------------------------------------------------
// fp32 -> bf16 elementwise convert (weights; 32MB bf16 total -> L2 resident)
// ---------------------------------------------------------------------------
__global__ void cvt_f32_bf16(const float* __restrict__ src, bf16* __restrict__ dst, int n) {
    int i = blockIdx.x * blockDim.x + threadIdx.x;
    if (i < n) dst[i] = (bf16)src[i];
}

// ---------------------------------------------------------------------------
// RMSNorm: out_bf16[row] = x[row] * rsqrt(mean(x^2)+eps) * g
// ---------------------------------------------------------------------------
__global__ __launch_bounds__(256) void rmsnorm_bf16_kernel(
    const float* __restrict__ x, const float* __restrict__ g,
    bf16* __restrict__ out, int d) {
    int row = blockIdx.x;
    const float* xr = x + (size_t)row * d;
    float ss = 0.f;
    for (int i = threadIdx.x; i < d; i += blockDim.x) { float v = xr[i]; ss += v * v; }
    __shared__ float red[32];
#pragma unroll
    for (int off = 16; off > 0; off >>= 1) ss += __shfl_xor(ss, off, 32);
    int wid = threadIdx.x >> 5, lid = threadIdx.x & 31;
    if (lid == 0) red[wid] = ss;
    __syncthreads();
    if (wid == 0) {
        float v = (lid < (int)(blockDim.x >> 5)) ? red[lid] : 0.f;
#pragma unroll
        for (int off = 16; off > 0; off >>= 1) v += __shfl_xor(v, off, 32);
        if (lid == 0) red[0] = v;
    }
    __syncthreads();
    float inv = rsqrtf(red[0] / (float)d + 1e-6f);
    for (int i = threadIdx.x; i < d; i += blockDim.x)
        out[(size_t)row * d + i] = (bf16)(xr[i] * inv * g[i]);
}

// ---------------------------------------------------------------------------
// Generic GEMM:  C[M,N] = A[M,K](bf16) * W[N,K](bf16)^T   (weight row = out feature)
// MODE 0: store bf16.
// MODE 1: store f32 = residual + acc.
// MODE 2: QKV epilogue — q,k cols (<2048) stored bf16 row-major; v cols (>=2048)
//         scattered directly into transposed layout vt[((b*16+h)*64+d)*2048 + t],
//         eliminating the separate V-transpose pass.
// Block 256 thr = 8 waves (4M x 2N). Block tile 128x128, wave tile 32x64.
// ---------------------------------------------------------------------------
template <int MODE>
__global__ __launch_bounds__(256) void gemm_bf16_kernel(
    const bf16* __restrict__ A, const bf16* __restrict__ W,
    const float* __restrict__ res, float* __restrict__ outf,
    bf16* __restrict__ outb, bf16* __restrict__ vtp,
    int M, int N, int K) {
    const int wave = threadIdx.x >> 5;
    const int lane = threadIdx.x & 31;
    const int half = lane >> 4;
    const int l16  = lane & 15;
    const int m0 = blockIdx.y * 128 + (wave >> 1) * 32;
    const int n0 = blockIdx.x * 128 + (wave & 1) * 64;

    v8f acc[2][4] = {};
    const bf16* a0 = A + (size_t)(m0 + l16) * K + 8 * half;
    const bf16* a1 = a0 + (size_t)16 * K;
    const bf16* w0 = W + (size_t)(n0 + l16) * K + 8 * half;

    for (int k = 0; k < K; k += 32) {
        v16bf af0 = load_frag(a0 + k);
        v16bf af1 = load_frag(a1 + k);
#pragma unroll
        for (int nf = 0; nf < 4; ++nf) {
            v16bf bfr = load_frag(w0 + (size_t)(nf * 16) * K + k);
            acc[0][nf] = wmma_bf(af0, bfr, acc[0][nf]);
            acc[1][nf] = wmma_bf(af1, bfr, acc[1][nf]);
        }
    }
#pragma unroll
    for (int mf = 0; mf < 2; ++mf)
#pragma unroll
        for (int nf = 0; nf < 4; ++nf)
#pragma unroll
            for (int r = 0; r < 8; ++r) {
                int row = m0 + mf * 16 + r + 8 * half;   // C layout: M = r + 8*(lane>>4)
                int col = n0 + nf * 16 + l16;            //           N = lane & 15
                size_t idx = (size_t)row * N + col;
                float v = acc[mf][nf][r];
                if (MODE == 0) {
                    outb[idx] = (bf16)v;
                } else if (MODE == 1) {
                    outf[idx] = v + res[idx];
                } else {                                  // MODE 2 (QKV)
                    if (col < 2048) {
                        outb[idx] = (bf16)v;              // q, k: row-major
                    } else {                              // v: transposed [b,h,d,t]
                        int d  = col & 63;
                        int hh = (col >> 6) & 15;
                        int bb = row >> 11;
                        int tt = row & 2047;
                        vtp[(size_t)((bb * 16 + hh) * 64 + d) * 2048 + tt] = (bf16)v;
                    }
                }
            }
}

// ---------------------------------------------------------------------------
// Dual GEMM + SwiGLU: S = silu(A*Wg^T) * (A*Wu^T), store bf16.
// Block 256 thr = 8 waves (2M x 4N). Block tile 64x128, wave tile 32x32.
// ---------------------------------------------------------------------------
__global__ __launch_bounds__(256) void gemm_swiglu_kernel(
    const bf16* __restrict__ A, const bf16* __restrict__ Wg,
    const bf16* __restrict__ Wu, bf16* __restrict__ S, int M, int N, int K) {
    const int wave = threadIdx.x >> 5;
    const int lane = threadIdx.x & 31;
    const int half = lane >> 4;
    const int l16  = lane & 15;
    const int m0 = blockIdx.y * 64  + (wave >> 2) * 32;
    const int n0 = blockIdx.x * 128 + (wave & 3) * 32;

    v8f ag[2][2] = {}, au[2][2] = {};
    const bf16* a0 = A  + (size_t)(m0 + l16) * K + 8 * half;
    const bf16* a1 = a0 + (size_t)16 * K;
    const bf16* g0 = Wg + (size_t)(n0 + l16) * K + 8 * half;
    const bf16* u0 = Wu + (size_t)(n0 + l16) * K + 8 * half;

    for (int k = 0; k < K; k += 32) {
        v16bf af0 = load_frag(a0 + k);
        v16bf af1 = load_frag(a1 + k);
#pragma unroll
        for (int nf = 0; nf < 2; ++nf) {
            v16bf gf = load_frag(g0 + (size_t)(nf * 16) * K + k);
            v16bf uf = load_frag(u0 + (size_t)(nf * 16) * K + k);
            ag[0][nf] = wmma_bf(af0, gf, ag[0][nf]);
            ag[1][nf] = wmma_bf(af1, gf, ag[1][nf]);
            au[0][nf] = wmma_bf(af0, uf, au[0][nf]);
            au[1][nf] = wmma_bf(af1, uf, au[1][nf]);
        }
    }
#pragma unroll
    for (int mf = 0; mf < 2; ++mf)
#pragma unroll
        for (int nf = 0; nf < 2; ++nf)
#pragma unroll
            for (int r = 0; r < 8; ++r) {
                float g = ag[mf][nf][r];
                float u = au[mf][nf][r];
                float s = (g / (1.f + __expf(-g))) * u;   // silu(g)*u
                int row = m0 + mf * 16 + r + 8 * half;
                int col = n0 + nf * 16 + l16;
                S[(size_t)row * N + col] = (bf16)s;
            }
}

// ---------------------------------------------------------------------------
// Flash attention, causal. One wave per (b, h, 32-query tile): two 16-row
// M-tiles share every K/V fragment (halves K/V L2 traffic; 16 WMMAs per
// 16x16B K/V loads in the inner loop).
// S = Q K^T via WMMA (K rows contiguous in d => natural B-fragments).
// Online softmax with cross-lane (16-lane half) reductions.
// P restaged through LDS into A-fragment layout, then P*V via WMMA on V^T.
// ---------------------------------------------------------------------------
__global__ __launch_bounds__(32) void attn_kernel(
    const bf16* __restrict__ qkv, const bf16* __restrict__ vt,
    bf16* __restrict__ y) {
    const int h  = blockIdx.y;
    const int b  = blockIdx.z;
    const int lane = threadIdx.x & 31;
    const int half = lane >> 4;
    const int l16  = lane & 15;
    const int q0 = blockIdx.x * 32;          // 32-query tile

    const size_t tokbase = (size_t)b * 2048;
    const bf16* Q  = qkv + (tokbase + q0) * 3072 + h * 64;
    const bf16* Kp = qkv + tokbase * 3072 + 1024 + h * 64;
    const bf16* Vt = vt + (size_t)(b * 16 + h) * 64 * 2048;

    v16bf qf[2][2];
#pragma unroll
    for (int mt = 0; mt < 2; ++mt) {
        const bf16* qp = Q + (size_t)(mt * 16 + l16) * 3072 + 8 * half;
        qf[mt][0] = load_frag(qp);        // d = 0..31
        qf[mt][1] = load_frag(qp + 32);   // d = 32..63
    }

    v8f acc[2][4] = {};
    float mrun[2][8], lrun[2][8];
#pragma unroll
    for (int mt = 0; mt < 2; ++mt)
#pragma unroll
        for (int r = 0; r < 8; ++r) { mrun[mt][r] = -3.0e38f; lrun[mt][r] = 0.f; }

    __shared__ alignas(16) bf16 plds[32][32];
    const float scale = 0.125f;     // 1/sqrt(64)
    const int kend = q0 + 32;       // exclusive key bound (causal)

    for (int kb = 0; kb < kend; kb += 32) {
        // --- scores: 2 M-tiles x 2 key subtiles, K fragments shared ---
        v8f s[2][2];                // [mt][sub]
#pragma unroll
        for (int sub = 0; sub < 2; ++sub) {
            const bf16* kp = Kp + (size_t)(kb + sub * 16 + l16) * 3072 + 8 * half;
            v16bf kf0 = load_frag(kp);
            v16bf kf1 = load_frag(kp + 32);
#pragma unroll
            for (int mt = 0; mt < 2; ++mt) {
                v8f st = {};
                st = wmma_bf(qf[mt][0], kf0, st);
                st = wmma_bf(qf[mt][1], kf1, st);
                s[mt][sub] = st;
            }
        }
        const bool need_mask = (kb + 31) > q0;
#pragma unroll
        for (int mt = 0; mt < 2; ++mt)
#pragma unroll
            for (int sub = 0; sub < 2; ++sub)
#pragma unroll
                for (int r = 0; r < 8; ++r) {
                    float v = s[mt][sub][r] * scale;
                    if (need_mask) {
                        int key  = kb + sub * 16 + l16;
                        int qrow = q0 + mt * 16 + r + 8 * half;
                        if (key > qrow) v = -3.0e38f;
                    }
                    s[mt][sub][r] = v;
                }
        // --- online softmax per M-tile ---
#pragma unroll
        for (int mt = 0; mt < 2; ++mt) {
            float mnew[8];
#pragma unroll
            for (int r = 0; r < 8; ++r) {
                float v = fmaxf(s[mt][0][r], s[mt][1][r]);
                v = fmaxf(v, __shfl_xor(v, 1, 32));
                v = fmaxf(v, __shfl_xor(v, 2, 32));
                v = fmaxf(v, __shfl_xor(v, 4, 32));
                v = fmaxf(v, __shfl_xor(v, 8, 32));
                mnew[r] = fmaxf(mrun[mt][r], v);
            }
#pragma unroll
            for (int sub = 0; sub < 2; ++sub)
#pragma unroll
                for (int r = 0; r < 8; ++r) {
                    float p = __expf(s[mt][sub][r] - mnew[r]);
                    s[mt][sub][r] = p;
                    plds[mt * 16 + r + 8 * half][sub * 16 + l16] = (bf16)p;
                }
#pragma unroll
            for (int r = 0; r < 8; ++r) {
                float rs = s[mt][0][r] + s[mt][1][r];
                rs += __shfl_xor(rs, 1, 32);
                rs += __shfl_xor(rs, 2, 32);
                rs += __shfl_xor(rs, 4, 32);
                rs += __shfl_xor(rs, 8, 32);
                float alpha = __expf(mrun[mt][r] - mnew[r]);
                lrun[mt][r] = lrun[mt][r] * alpha + rs;
                mrun[mt][r] = mnew[r];
#pragma unroll
                for (int nt = 0; nt < 4; ++nt) acc[mt][nt][r] *= alpha;
            }
        }
        __syncthreads();
        // P as A-fragments: row = mt*16 + l16, K chunks at 8*half and 8*half+16
        v16bf pf0 = load_frag(&plds[l16][8 * half]);
        v16bf pf1 = load_frag(&plds[16 + l16][8 * half]);
        __syncthreads();
        // --- acc += P (32x32) * V (32x64); V fragments shared by both M-tiles ---
#pragma unroll
        for (int nt = 0; nt < 4; ++nt) {
            const bf16* vp = Vt + (size_t)(nt * 16 + l16) * 2048 + kb + 8 * half;
            v16bf vf = load_frag(vp);
            acc[0][nt] = wmma_bf(pf0, vf, acc[0][nt]);
            acc[1][nt] = wmma_bf(pf1, vf, acc[1][nt]);
        }
    }
    // --- normalize + store y (bf16, feeds out-proj GEMM) ---
    bf16* yout = y + (tokbase + q0) * 1024 + h * 64;
#pragma unroll
    for (int mt = 0; mt < 2; ++mt)
#pragma unroll
        for (int nt = 0; nt < 4; ++nt)
#pragma unroll
            for (int r = 0; r < 8; ++r) {
                int row = mt * 16 + r + 8 * half;
                int col = nt * 16 + l16;
                yout[(size_t)row * 1024 + col] = (bf16)(acc[mt][nt][r] / lrun[mt][r]);
            }
}

// ---------------------------------------------------------------------------
// Orchestration
// ---------------------------------------------------------------------------
extern "C" void kernel_launch(void* const* d_in, const int* in_sizes, int n_in,
                              void* d_out, int out_size, void* d_ws, size_t ws_size,
                              hipStream_t stream) {
    const float* x    = (const float*)d_in[0];
    const float* wqkv = (const float*)d_in[1];
    const float* wout = (const float*)d_in[2];
    const float* g1   = (const float*)d_in[3];
    const float* g2   = (const float*)d_in[4];
    const float* wg   = (const float*)d_in[5];
    const float* wu   = (const float*)d_in[6];
    const float* wo   = (const float*)d_in[7];
    float* out = (float*)d_out;
    char* ws = (char*)d_ws;

    // workspace layout (bytes), ~96 MB total with reuse
    bf16* wqkv_b = (bf16*)(ws + 0);          //  6,291,456
    bf16* wout_b = (bf16*)(ws + 6291456);    //  2,097,152
    bf16* wg_b   = (bf16*)(ws + 8388608);    //  8,388,608
    bf16* wu_b   = (bf16*)(ws + 16777216);   //  8,388,608
    bf16* wo_b   = (bf16*)(ws + 25165824);   //  8,388,608
    bf16* hb     = (bf16*)(ws + 33554432);   //  8,388,608 (h, reused as h2)
    bf16* qkv_b  = (bf16*)(ws + 41943040);   // 25,165,824 (q,k rows; v region unused)
    bf16* vt     = (bf16*)(ws + 67108864);   //  8,388,608 (V transposed [b,h,d,t])
    bf16* sb     = (bf16*)(ws + 41943040);   // 33,554,432 (reuses qkv+vt after attn)
    bf16* yb     = (bf16*)(ws + 75497472);   //  8,388,608
    float* x1    = (float*)(ws + 83886080);  // 16,777,216

    // weights -> bf16 (L2-resident thereafter)
    cvt_f32_bf16<<<3145728 / 256, 256, 0, stream>>>(wqkv, wqkv_b, 3145728);
    cvt_f32_bf16<<<1048576 / 256, 256, 0, stream>>>(wout, wout_b, 1048576);
    cvt_f32_bf16<<<4194304 / 256, 256, 0, stream>>>(wg,   wg_b,   4194304);
    cvt_f32_bf16<<<4194304 / 256, 256, 0, stream>>>(wu,   wu_b,   4194304);
    cvt_f32_bf16<<<4194304 / 256, 256, 0, stream>>>(wo,   wo_b,   4194304);

    // h = rmsnorm(x, g1)
    rmsnorm_bf16_kernel<<<4096, 256, 0, stream>>>(x, g1, hb, 1024);
    // qkv = h @ w_qkv^T [4096 x 3072]; V scattered transposed in-epilogue
    gemm_bf16_kernel<2><<<dim3(3072 / 128, 4096 / 128), 256, 0, stream>>>(
        hb, wqkv_b, nullptr, nullptr, qkv_b, vt, 4096, 3072, 1024);
    // causal flash attention -> y
    attn_kernel<<<dim3(64, 16, 2), 32, 0, stream>>>(qkv_b, vt, yb);
    // x1 = x + y @ w_out^T
    gemm_bf16_kernel<1><<<dim3(1024 / 128, 4096 / 128), 256, 0, stream>>>(
        yb, wout_b, x, x1, nullptr, nullptr, 4096, 1024, 1024);
    // h2 = rmsnorm(x1, g2)
    rmsnorm_bf16_kernel<<<4096, 256, 0, stream>>>(x1, g2, hb, 1024);
    // s = silu(h2 @ w_g^T) * (h2 @ w_u^T)   [4096 x 4096]
    gemm_swiglu_kernel<<<dim3(4096 / 128, 4096 / 64), 256, 0, stream>>>(
        hb, wg_b, wu_b, sb, 4096, 4096, 1024);
    // out = x1 + s @ w_o^T
    gemm_bf16_kernel<1><<<dim3(1024 / 128, 4096 / 128), 256, 0, stream>>>(
        sb, wo_b, x1, out, nullptr, nullptr, 4096, 1024, 4096);
}
